// spectralNNShallow_38362647888529
// MI455X (gfx1250) — matrix operand order repaired and checked by reading, not compiled, not run on hardware
//
#include <hip/hip_runtime.h>
#include <hip/hip_bf16.h>

typedef __attribute__((ext_vector_type(16))) _Float16 v16h;
typedef __attribute__((ext_vector_type(8)))  float    v8f;
typedef __attribute__((ext_vector_type(4)))  unsigned int v4u;
typedef __attribute__((ext_vector_type(8)))  int          v8i_t;
typedef __attribute__((ext_vector_type(4)))  int          v4i_t;

#define M_CNT     256
#define W_WIN     33
#define WP        64          // padded window (zeros for j >= 33)
#define D_DIM     512
#define B_COLS    1024
#define N_ROWS    4096
#define XI_STRIDE 4128        // N + 2L

// ---------------------------------------------------------------------------
// TDM: async 2D tile load Global->LDS via Tensor Data Mover (D# per ISA ch.8).
// tile_d0 elements (2B each) per row, tile_d1 rows, row stride = stride_elems.
// ---------------------------------------------------------------------------
__device__ __forceinline__ void tdm_load_2d_f16(unsigned lds_off, const void* gaddr,
                                                unsigned tile_d0, unsigned tile_d1,
                                                unsigned stride_elems) {
  const unsigned long long ga = (unsigned long long)gaddr;
  const unsigned td0 = 0x40000000u;   // huge tensor dims -> no OOB clipping
  const unsigned td1 = 0x40000000u;
  v4u g0;
  g0[0] = 1u;                                          // count=1, user mode
  g0[1] = lds_off;                                     // lds_addr (bytes)
  g0[2] = (unsigned)(ga & 0xffffffffu);                // global_addr[31:0]
  g0[3] = (unsigned)((ga >> 32) & 0x01ffffffu)         // global_addr[56:32]
          | (2u << 30);                                // type = 2 ("image")
  v8i_t g1;
  g1[0] = (int)(1u << 16);                             // wg_mask=0, data_size=1 (2B)
  g1[1] = (int)((td0 & 0xffffu) << 16);                // [63:48] = tensor_dim0 lo16
  g1[2] = (int)(((td0 >> 16) & 0xffffu) | ((td1 & 0xffffu) << 16));
  g1[3] = (int)(((td1 >> 16) & 0xffffu) | (tile_d0 << 16));  // tile_dim0
  g1[4] = (int)(tile_d1 & 0xffffu);                    // tile_dim1 (tile_dim2=0)
  g1[5] = (int)stride_elems;                           // tensor_dim0_stride lo32
  g1[6] = 0;
  g1[7] = 0;
  const v4i_t z4 = {0, 0, 0, 0};
#if defined(__clang_major__) && __clang_major__ >= 23
  const v8i_t z8 = {0, 0, 0, 0, 0, 0, 0, 0};
  __builtin_amdgcn_tensor_load_to_lds(g0, g1, z4, z4, z8, 0);
#else
  __builtin_amdgcn_tensor_load_to_lds(g0, g1, z4, z4, 0);
#endif
}

// ---------------------------------------------------------------------------
// Kernel 1: Gpad[m, j, b] = sigmoid( sum_d weight[m,j,d] * u[b,d] + bias[m,j] )
// GEMM (8448 x 512) x (512 x 1024), f16 WMMA, f32 accumulate.
// ---------------------------------------------------------------------------
__global__ __launch_bounds__(128) void spectral_gate_gemm(
    const float* __restrict__ u, const float* __restrict__ weight,
    const float* __restrict__ bias, _Float16* __restrict__ Gpad)
{
  const int lane  = threadIdx.x & 31;
  const int wid   = threadIdx.x >> 5;
  const int rt    = blockIdx.x;              // 0..527   row tile (16 rows of M*W)
  const int ct    = blockIdx.y * 4 + wid;    // 0..63    col tile (16 cols of B)
  const int mj    = rt * 16 + (lane & 15);   // A row = weight row
  const int halfK = (lane < 16) ? 0 : 8;

  v8f acc = {0.f, 0.f, 0.f, 0.f, 0.f, 0.f, 0.f, 0.f};

  for (int kc = 0; kc < 16; ++kc) {
    const int kbase = kc << 5;
    v16h a, b;
#pragma unroll
    for (int vv = 0; vv < 8; ++vv) {
      const int K = kbase + ((vv >> 2) << 4) + halfK + ((vv & 3) << 1); // even
      const float2 w2 = *reinterpret_cast<const float2*>(&weight[mj * D_DIM + K]);
      a[2 * vv]     = (_Float16)w2.x;
      a[2 * vv + 1] = (_Float16)w2.y;
    }
    const int ncol = ct * 16 + (lane & 15);  // B: lane = column, elements = K
#pragma unroll
    for (int vv = 0; vv < 8; ++vv) {
      const int K = kbase + ((vv >> 2) << 4) + halfK + ((vv & 3) << 1); // even
      b[2 * vv]     = (_Float16)u[ncol * D_DIM + K];
      b[2 * vv + 1] = (_Float16)u[ncol * D_DIM + K + 1];
    }
    acc = __builtin_amdgcn_wmma_f32_16x16x32_f16(false, a, false, b,
                                                 (short)0, acc, false, false);
  }

  const int col   = lane & 15;
  const int rbase = (lane >> 4) << 3;
#pragma unroll
  for (int vv = 0; vv < 8; ++vv) {
    const int mjr = rt * 16 + vv + rbase;
    float g = acc[vv] + bias[mjr];
    g = 1.0f / (1.0f + __expf(-g));
    const int m = mjr / W_WIN;
    const int j = mjr - m * W_WIN;
    Gpad[(m * WP + j) * B_COLS + ct * 16 + col] = (_Float16)g;
  }
}

// ---------------------------------------------------------------------------
// Kernel 2: out[n, b] = sum_m sum_j xi[m, n+j] * G[m, j, b]
// Toeplitz A built in LDS; G tiles streamed by the TDM, double-buffered.
// Block: 256 threads = 8 waves; 128(n) x 64(b) output tile per block.
// ---------------------------------------------------------------------------
__global__ __launch_bounds__(256) void spectral_apply_gemm(
    const float* __restrict__ xi, const _Float16* __restrict__ Gpad,
    float* __restrict__ out)
{
  __shared__ _Float16 Atile[128 * 64];      // 16 KB: A[r][j] = xi[m, n0+r+j]
  __shared__ _Float16 Gtile[2][64 * 64];    // 2 x 8 KB: G[j][b-b0], double buffer

  const int t    = threadIdx.x;
  const int lane = t & 31;
  const int wid  = t >> 5;
  const int b0   = blockIdx.x * 64;
  const int n0   = blockIdx.y * 128;

  v8f acc[4];
#pragma unroll
  for (int c = 0; c < 4; ++c) acc[c] = (v8f){0.f,0.f,0.f,0.f,0.f,0.f,0.f,0.f};

  const int halfK = (lane < 16) ? 0 : 8;
  const int arow  = wid * 16 + (lane & 15);

  // Prologue: kick off TDM for m=0 into buffer 0 (wave 0 only; TDM ignores EXEC
  // but the uniform branch skips the whole instruction for waves 1..7).
  if (t < 32)
    tdm_load_2d_f16((unsigned)(unsigned long long)&Gtile[0][0],
                    &Gpad[b0], 64, 64, B_COLS);

  for (int m = 0; m < M_CNT; ++m) {
    // --- stage Toeplitz A tile (clamped reads only ever multiply zero G rows)
    const float* xim = xi + m * XI_STRIDE;
#pragma unroll
    for (int i = 0; i < 16; ++i) {
      const int p  = t + i * 256;          // 0..4095
      const int rr = p >> 5;               // row 0..127
      const int jp = (p & 31) << 1;        // even j 0..62
      int i0 = n0 + rr + jp;  if (i0 > XI_STRIDE - 1) i0 = XI_STRIDE - 1;
      int i1 = i0 + 1;        if (i1 > XI_STRIDE - 1) i1 = XI_STRIDE - 1;
      union { _Float16 h[2]; unsigned w; } pk;
      pk.h[0] = (_Float16)xim[i0];
      pk.h[1] = (_Float16)xim[i1];
      *reinterpret_cast<unsigned*>(&Atile[rr * 64 + jp]) = pk.w;
    }
    // --- prefetch next G tile via TDM into the other buffer
    if (t < 32 && m + 1 < M_CNT)
      tdm_load_2d_f16((unsigned)(unsigned long long)&Gtile[(m + 1) & 1][0],
                      &Gpad[(size_t)(m + 1) * WP * B_COLS + b0], 64, 64, B_COLS);
    // TDM completes in-order per wave: <=1 outstanding means current tile landed.
    if (m + 1 < M_CNT) __builtin_amdgcn_s_wait_tensorcnt((short)1);
    else               __builtin_amdgcn_s_wait_tensorcnt((short)0);
    __syncthreads();

    const _Float16* Gt = &Gtile[m & 1][0];
#pragma unroll
    for (int chunk = 0; chunk < 2; ++chunk) {
      const int kbase = chunk << 5;
      union { v16h h; unsigned w[8]; } af;
#pragma unroll
      for (int vv = 0; vv < 8; ++vv) {
        const int K = kbase + ((vv >> 2) << 4) + halfK + ((vv & 3) << 1); // even
        af.w[vv] = *reinterpret_cast<const unsigned*>(&Atile[arow * 64 + K]);
      }
      const int bcol = lane & 15;          // column within 16-wide G subtile
#pragma unroll
      for (int c = 0; c < 4; ++c) {
        union { v16h h; unsigned w[8]; } bf;
#pragma unroll
        for (int vv = 0; vv < 8; ++vv) {
          const int K = kbase + ((vv >> 2) << 4) + halfK + ((vv & 3) << 1);
          bf.h[2 * vv]     = Gt[K * 64 + c * 16 + bcol];
          bf.h[2 * vv + 1] = Gt[(K + 1) * 64 + c * 16 + bcol];
        }
        acc[c] = __builtin_amdgcn_wmma_f32_16x16x32_f16(
            false, af.h, false, bf.h, (short)0, acc[c], false, false);
      }
    }
    __syncthreads();
  }

  const int col = lane & 15;
  const int rb  = (lane >> 4) << 3;
#pragma unroll
  for (int c = 0; c < 4; ++c)
#pragma unroll
    for (int vv = 0; vv < 8; ++vv)
      out[(n0 + wid * 16 + vv + rb) * B_COLS + b0 + c * 16 + col] = acc[c][vv];
}

extern "C" void kernel_launch(void* const* d_in, const int* in_sizes, int n_in,
                              void* d_out, int out_size, void* d_ws, size_t ws_size,
                              hipStream_t stream) {
  const float* u      = (const float*)d_in[0];  // (1024, 512)
  const float* weight = (const float*)d_in[1];  // (256, 33, 512)
  const float* bias   = (const float*)d_in[2];  // (256, 33, 1)
  const float* xi     = (const float*)d_in[3];  // (256, 4128)
  float*       out    = (float*)d_out;          // (4096, 1024)
  _Float16*    Gpad   = (_Float16*)d_ws;        // (256, 64, 1024) f16, 33.5 MB

  // zero-fill so padded window rows (j >= 33) contribute nothing
  hipMemsetAsync(d_ws, 0, (size_t)M_CNT * WP * B_COLS * sizeof(_Float16), stream);

  spectral_gate_gemm<<<dim3(528, 16), 128, 0, stream>>>(u, weight, bias, Gpad);
  spectral_apply_gemm<<<dim3(16, 32), 256, 0, stream>>>(xi, Gpad, out);
}
